// EmbeddingForce_9156870275571
// MI455X (gfx1250) — matrix operand order repaired and checked by reading, not compiled, not run on hardware
//
#include <hip/hip_runtime.h>

#define DMODEL 1280
#define NHEAD  4
#define DHEAD  320
#define FFDIM  256
#define NLAYER 2
#define NGRAPH 64
#define NTOK   32768
#define LMAX   640
#define MPAD   (NGRAPH * LMAX)   /* 40960 rows, multiple of 128 */

typedef unsigned short u16;
typedef __attribute__((ext_vector_type(16))) __bf16 v16bf;
typedef __attribute__((ext_vector_type(8)))  float  v8f;

// ---------- helpers ----------
__device__ inline u16 f2bf(float f) {
  unsigned int u = __float_as_uint(f);
  u += 0x7FFFu + ((u >> 16) & 1u);          // round-to-nearest-even
  return (u16)(u >> 16);
}

__device__ inline v8f wmma_bf16(uint4 a0, uint4 a1, uint4 b0, uint4 b1, v8f c) {
  union { v16bf v; uint4 q[2]; } A, B;
  A.q[0] = a0; A.q[1] = a1; B.q[0] = b0; B.q[1] = b1;
  return __builtin_amdgcn_wmma_f32_16x16x32_bf16(false, A.v, false, B.v,
                                                 (short)0, c, false, false);
}

// async copy of one 16B chunk, global -> LDS, tracked by ASYNCcnt (CDNA5 TDM-lite path)
__device__ inline void async_cp16(unsigned lds_off, unsigned byte_off,
                                  unsigned long long base) {
  asm volatile("global_load_async_to_lds_b128 %0, %1, %2"
               :: "v"(lds_off), "v"(byte_off), "s"(base)
               : "memory");
}

__device__ inline float redmax16(float v) {
  v = fmaxf(v, __shfl_xor(v, 1, 32));
  v = fmaxf(v, __shfl_xor(v, 2, 32));
  v = fmaxf(v, __shfl_xor(v, 4, 32));
  v = fmaxf(v, __shfl_xor(v, 8, 32));
  return v;
}
__device__ inline float redsum16(float v) {
  v += __shfl_xor(v, 1, 32);
  v += __shfl_xor(v, 2, 32);
  v += __shfl_xor(v, 4, 32);
  v += __shfl_xor(v, 8, 32);
  return v;
}

// ---------- small utility kernels ----------
__global__ void k_zero_f32(float* p, size_t n) {
  size_t i = (size_t)blockIdx.x * blockDim.x + threadIdx.x;
  if (i < n) p[i] = 0.f;
}
__global__ void k_cvt_bf16(const float* __restrict__ s, u16* __restrict__ d, size_t n) {
  size_t i = (size_t)blockIdx.x * blockDim.x + threadIdx.x;
  if (i < n) d[i] = f2bf(s[i]);
}
__global__ void k_zero_counts(int* counts) { counts[threadIdx.x] = 0; }
__global__ void k_count(const int* __restrict__ batch, int* counts) {
  int i = blockIdx.x * blockDim.x + threadIdx.x;
  if (i < NTOK) atomicAdd(&counts[batch[i]], 1);
}
__global__ void k_scan(const int* __restrict__ counts, int* offsets) {
  if (threadIdx.x == 0) {
    int acc = 0;
    for (int b = 0; b < NGRAPH; ++b) { offsets[b] = acc; acc += counts[b]; }
  }
}
__global__ void k_pos(const int* __restrict__ batch, const int* __restrict__ offsets,
                      int* __restrict__ pos) {
  int i = blockIdx.x * blockDim.x + threadIdx.x;
  if (i < NTOK) pos[i] = i - offsets[batch[i]];
}
__global__ void k_scatter(const float* __restrict__ x, const int* __restrict__ batch,
                          const int* __restrict__ pos, float* __restrict__ h) {
  size_t idx = (size_t)blockIdx.x * blockDim.x + threadIdx.x;
  if (idx >= (size_t)NTOK * DMODEL) return;
  int i = (int)(idx / DMODEL), d = (int)(idx % DMODEL);
  int p = pos[i];
  if (p < LMAX) h[((size_t)batch[i] * LMAX + p) * DMODEL + d] = x[idx];
}
__global__ void k_gather(const float* __restrict__ y, const int* __restrict__ batch,
                         const int* __restrict__ pos, float* __restrict__ out) {
  size_t idx = (size_t)blockIdx.x * blockDim.x + threadIdx.x;
  if (idx >= (size_t)NTOK * DMODEL) return;
  int i = (int)(idx / DMODEL), d = (int)(idx % DMODEL);
  int p = pos[i];
  out[idx] = (p < LMAX) ? y[((size_t)batch[i] * LMAX + p) * DMODEL + d] : 0.f;
}

// coalesced 32x32 LDS tile transpose: vt[b][h][dh][l] = V slice of qkv
__global__ __launch_bounds__(256) void k_vtrans(const u16* __restrict__ qkv,
                                                u16* __restrict__ vt) {
  __shared__ u16 t[32][33];
  int bid = blockIdx.x;                         // (B*H) x (DH/32) x (L/32)
  int tl  = bid % (LMAX / 32);   bid /= (LMAX / 32);
  int td  = bid % (DHEAD / 32);  bid /= (DHEAD / 32);
  int hh  = bid % NHEAD;
  int b   = bid / NHEAD;
  const int l0 = tl * 32, d0 = td * 32;
  const int c = threadIdx.x & 31, r = threadIdx.x >> 5;
#pragma unroll
  for (int i = 0; i < 4; ++i) {                 // read: consecutive lanes along dh
    int lrow = r + i * 8;
    t[c][lrow] = qkv[((size_t)(b * LMAX + l0 + lrow)) * (3 * DMODEL) +
                     2 * DMODEL + hh * DHEAD + d0 + c];
  }
  __syncthreads();
#pragma unroll
  for (int i = 0; i < 4; ++i) {                 // write: consecutive lanes along l
    int drow = r + i * 8;
    vt[((size_t)((b * NHEAD + hh) * DHEAD + d0 + drow)) * LMAX + l0 + c] = t[drow][c];
  }
}

// ---------- fused residual + LayerNorm (+bf16 recast) ----------
__global__ __launch_bounds__(256) void k_ln(float* __restrict__ h,
                                            const float* __restrict__ y,
                                            const float* __restrict__ w,
                                            const float* __restrict__ b,
                                            u16* __restrict__ hb) {
  __shared__ float s1[256], s2[256];
  const int row = blockIdx.x, tid = threadIdx.x;
  float vloc[5], s = 0.f, sq = 0.f;
#pragma unroll
  for (int i = 0; i < 5; ++i) {
    int d = tid + i * 256;
    float v = h[(size_t)row * DMODEL + d] + y[(size_t)row * DMODEL + d];
    vloc[i] = v; s += v; sq += v * v;
  }
  s1[tid] = s; s2[tid] = sq; __syncthreads();
  for (int off = 128; off > 0; off >>= 1) {
    if (tid < off) { s1[tid] += s1[tid + off]; s2[tid] += s2[tid + off]; }
    __syncthreads();
  }
  float mu = s1[0] * (1.0f / DMODEL);
  float var = s2[0] * (1.0f / DMODEL) - mu * mu;
  float rstd = rsqrtf(var + 1e-5f);
#pragma unroll
  for (int i = 0; i < 5; ++i) {
    int d = tid + i * 256;
    float o = (vloc[i] - mu) * rstd * w[d] + b[d];
    h[(size_t)row * DMODEL + d]  = o;
    hb[(size_t)row * DMODEL + d] = f2bf(o);
  }
}

// ---------- bf16 WMMA GEMM:  C = A[M,K] @ W[N,K]^T + bias ----------
// block tile 128x256, 8 waves (2x4), wave tile 64x64 (4x4 WMMA accumulators)
// double-buffered LDS fed by global_load_async_to_lds_b128 (ASYNCcnt pipeline)
__global__ __launch_bounds__(256) void k_gemm(const u16* __restrict__ A,
                                              const u16* __restrict__ W,
                                              const float* __restrict__ bias,
                                              float* __restrict__ Cf,
                                              u16* __restrict__ Cb,
                                              int Nout, int K, int relu) {
  __shared__ __align__(16) u16 lsA[2][128 * 40];   // row stride 40 -> 16B-aligned,
  __shared__ __align__(16) u16 lsB[2][256 * 40];   // conflict-free b128 frag loads
  const int tid  = threadIdx.x;
  const int lane = tid & 31, half = lane >> 4, l15 = lane & 15;
  const int wave = tid >> 5, wm = wave >> 2, wn = wave & 3;
  const int bm = blockIdx.y * 128, bn = blockIdx.x * 256;

  const unsigned ldsA0 = (unsigned)(uintptr_t)(void*)&lsA[0][0];
  const unsigned ldsB0 = (unsigned)(uintptr_t)(void*)&lsB[0][0];
  const unsigned long long baseA = (unsigned long long)(uintptr_t)A;
  const unsigned long long baseW = (unsigned long long)(uintptr_t)W;

  v8f acc[4][4];
#pragma unroll
  for (int mi = 0; mi < 4; ++mi)
#pragma unroll
    for (int ni = 0; ni < 4; ++ni)
#pragma unroll
      for (int e = 0; e < 8; ++e) acc[mi][ni][e] = 0.f;

  // issue one k-tile: 2 A chunks + 4 B chunks of 16B per thread (6 async ops/wave)
  auto issue = [&](int buf, int k0) {
#pragma unroll
    for (int i = 0; i < 2; ++i) {                  // A: 128x32 = 512 chunks
      int c = tid + i * 256, row = c >> 2, kc = c & 3;
      async_cp16(ldsA0 + (unsigned)(buf * 128 * 40 + row * 40 + kc * 8) * 2,
                 (unsigned)(((unsigned)(bm + row) * (unsigned)K + k0 + kc * 8) * 2),
                 baseA);
    }
#pragma unroll
    for (int i = 0; i < 4; ++i) {                  // B: 256x32 = 1024 chunks
      int c = tid + i * 256, row = c >> 2, kc = c & 3;
      async_cp16(ldsB0 + (unsigned)(buf * 256 * 40 + row * 40 + kc * 8) * 2,
                 (unsigned)(((unsigned)(bn + row) * (unsigned)K + k0 + kc * 8) * 2),
                 baseW);
    }
  };

  const int nk = K >> 5;
  issue(0, 0);
  for (int it = 0; it < nk; ++it) {
    if (it + 1 < nk) {
      issue((it + 1) & 1, (it + 1) << 5);
      asm volatile("s_wait_asynccnt 0x6" ::: "memory");   // current tile landed
    } else {
      asm volatile("s_wait_asynccnt 0x0" ::: "memory");
    }
    __syncthreads();
    const int buf = it & 1;
    uint4 a0[4], a1[4];
#pragma unroll
    for (int mi = 0; mi < 4; ++mi) {
      int mr = wm * 64 + mi * 16 + l15;
      a0[mi] = *(const uint4*)&lsA[buf][mr * 40 + half * 8];        // k = half*8..
      a1[mi] = *(const uint4*)&lsA[buf][mr * 40 + 16 + half * 8];   // k = 16+half*8..
    }
#pragma unroll
    for (int ni = 0; ni < 4; ++ni) {
      int nc = wn * 64 + ni * 16 + l15;
      uint4 b0 = *(const uint4*)&lsB[buf][nc * 40 + half * 8];      // k = half*8..
      uint4 b1 = *(const uint4*)&lsB[buf][nc * 40 + 16 + half * 8];
#pragma unroll
      for (int mi = 0; mi < 4; ++mi)
        acc[mi][ni] = wmma_bf16(a0[mi], a1[mi], b0, b1, acc[mi][ni]);
    }
    __syncthreads();
  }
#pragma unroll
  for (int ni = 0; ni < 4; ++ni) {
    int col = bn + wn * 64 + ni * 16 + l15;
    float bv = bias ? bias[col] : 0.f;
#pragma unroll
    for (int mi = 0; mi < 4; ++mi)
#pragma unroll
      for (int r = 0; r < 8; ++r) {
        int row = bm + wm * 64 + mi * 16 + r + half * 8;
        float v = acc[mi][ni][r] + bv;
        if (relu) v = fmaxf(v, 0.f);
        if (Cf) Cf[(size_t)row * Nout + col] = v;
        else    Cb[(size_t)row * Nout + col] = f2bf(v);
      }
  }
}

// ---------- flash attention: 1 wave per (graph, head, 16-row q tile) ----------
__global__ __launch_bounds__(32) void k_attn(const u16* __restrict__ qkv,
                                             const u16* __restrict__ vt,
                                             const int* __restrict__ counts,
                                             u16* __restrict__ out) {
  __shared__ __align__(16) u16 lp[16 * 32];
  const int bid = blockIdx.x;
  const int qt = bid % (LMAX / 16);
  const int hh = (bid / (LMAX / 16)) % NHEAD;
  const int b  = bid / ((LMAX / 16) * NHEAD);
  const int count = counts[b];
  const int q0 = qt * 16;
  if (q0 >= count) return;                          // pure-padding tile
  const int lane = threadIdx.x, half = lane >> 4, l15 = lane & 15;
  const float scale = 0.05590169943749474f;         // 1/sqrt(320)

  v8f O[20];
#pragma unroll
  for (int t = 0; t < 20; ++t)
#pragma unroll
    for (int e = 0; e < 8; ++e) O[t][e] = 0.f;
  float mrun[8], lrun[8];
#pragma unroll
  for (int r = 0; r < 8; ++r) { mrun[r] = -3e38f; lrun[r] = 0.f; }

  const u16* qbase = qkv + ((size_t)(b * LMAX + q0 + l15)) * (3 * DMODEL) + hh * DHEAD;
  int nch = (count + 31) >> 5; if (nch > LMAX / 32) nch = LMAX / 32;

  for (int kc = 0; kc < nch; ++kc) {
    const int kb = kc * 32;
    v8f S0, S1;
#pragma unroll
    for (int e = 0; e < 8; ++e) { S0[e] = 0.f; S1[e] = 0.f; }
#pragma unroll
    for (int kk = 0; kk < 10; ++kk) {               // Q @ K^T over DH=320
      uint4 a0 = *(const uint4*)(qbase + kk * 32 + half * 8);
      uint4 a1 = *(const uint4*)(qbase + kk * 32 + 16 + half * 8);
      const u16* kp0 = qkv + ((size_t)(b * LMAX + kb + l15)) * (3 * DMODEL) +
                       DMODEL + hh * DHEAD + kk * 32 + half * 8;
      const u16* kp1 = kp0 + (size_t)16 * (3 * DMODEL);
      S0 = wmma_bf16(a0, a1, *(const uint4*)kp0, *(const uint4*)(kp0 + 16), S0);
      S1 = wmma_bf16(a0, a1, *(const uint4*)kp1, *(const uint4*)(kp1 + 16), S1);
    }
    float alpha[8];
#pragma unroll
    for (int r = 0; r < 8; ++r) {                   // online softmax per row
      float s0 = S0[r] * scale, s1 = S1[r] * scale;
      if (kb + l15 >= count)      s0 = -3e38f;
      if (kb + 16 + l15 >= count) s1 = -3e38f;
      float mnew = fmaxf(mrun[r], redmax16(fmaxf(s0, s1)));
      float p0 = __expf(s0 - mnew), p1 = __expf(s1 - mnew);
      float rs = redsum16(p0 + p1);
      alpha[r] = __expf(mrun[r] - mnew);
      lrun[r] = lrun[r] * alpha[r] + rs;
      mrun[r] = mnew;
      lp[(r + 8 * half) * 32 + l15]      = f2bf(p0);  // C-layout -> LDS
      lp[(r + 8 * half) * 32 + 16 + l15] = f2bf(p1);
    }
#pragma unroll
    for (int t = 0; t < 20; ++t)
#pragma unroll
      for (int r = 0; r < 8; ++r) O[t][r] *= alpha[r];
    uint4 p0v = *(const uint4*)&lp[l15 * 32 + half * 8];     // P as A-fragment
    uint4 p1v = *(const uint4*)&lp[l15 * 32 + 16 + half * 8];
#pragma unroll
    for (int t = 0; t < 20; ++t) {                  // P @ V over 20 dh tiles
      const u16* vp = vt + ((size_t)((b * NHEAD + hh) * DHEAD + t * 16 + l15)) * LMAX +
                      kb + half * 8;
      O[t] = wmma_bf16(p0v, p1v, *(const uint4*)vp, *(const uint4*)(vp + 16), O[t]);
    }
  }
#pragma unroll
  for (int r = 0; r < 8; ++r) {
    float inv = 1.0f / lrun[r];
    int row = b * LMAX + q0 + r + 8 * half;
#pragma unroll
    for (int t = 0; t < 20; ++t)
      out[(size_t)row * DMODEL + hh * DHEAD + t * 16 + l15] = f2bf(O[t][r] * inv);
  }
}

// ---------- host ----------
extern "C" void kernel_launch(void* const* d_in, const int* in_sizes, int n_in,
                              void* d_out, int out_size, void* d_ws, size_t ws_size,
                              hipStream_t stream) {
  const float* x     = (const float*)d_in[0];
  const int*   batch = (const int*)d_in[1];
  const float* Wqkv  = (const float*)d_in[2];
  const float* bqkv  = (const float*)d_in[3];
  const float* Wo    = (const float*)d_in[4];
  const float* bo    = (const float*)d_in[5];
  const float* ln1w  = (const float*)d_in[6];
  const float* ln1b  = (const float*)d_in[7];
  const float* W1    = (const float*)d_in[8];
  const float* b1    = (const float*)d_in[9];
  const float* W2    = (const float*)d_in[10];
  const float* b2    = (const float*)d_in[11];
  const float* ln2w  = (const float*)d_in[12];
  const float* ln2b  = (const float*)d_in[13];
  const float* Wm1   = (const float*)d_in[14];
  const float* bm1   = (const float*)d_in[15];
  const float* Wm2   = (const float*)d_in[16];
  const float* bm2   = (const float*)d_in[17];
  float* outp = (float*)d_out;

  size_t off = 0;
  auto carve = [&](size_t bytes) -> char* {
    char* p = (char*)d_ws + off;
    off += (bytes + 255) & ~(size_t)255;
    return p;
  };
  int*   counts  = (int*)carve(NGRAPH * 4);
  int*   offsets = (int*)carve(NGRAPH * 4);
  int*   pos     = (int*)carve((size_t)NTOK * 4);
  float* h       = (float*)carve((size_t)MPAD * DMODEL * 4);
  u16*   hb      = (u16*)carve((size_t)MPAD * DMODEL * 2);
  float* y       = (float*)carve((size_t)MPAD * DMODEL * 4);
  u16*   vt      = (u16*)y;                       // phase-disjoint alias
  u16*   qkv     = (u16*)carve((size_t)MPAD * 3 * DMODEL * 2);
  u16*   ffb     = qkv;                           // phase-disjoint alias
  u16*   wq  = (u16*)carve((size_t)NLAYER * 3 * DMODEL * DMODEL * 2);
  u16*   wo  = (u16*)carve((size_t)NLAYER * DMODEL * DMODEL * 2);
  u16*   w1  = (u16*)carve((size_t)NLAYER * FFDIM * DMODEL * 2);
  u16*   w2  = (u16*)carve((size_t)NLAYER * DMODEL * FFDIM * 2);
  u16*   wm1 = (u16*)carve((size_t)FFDIM * DMODEL * 2);
  u16*   wm2 = (u16*)carve((size_t)DMODEL * FFDIM * 2);

  auto blks = [](size_t n, int t) { return (unsigned)((n + t - 1) / t); };
  auto gemm = [&](const u16* A, const u16* W, const float* bias,
                  float* Cf, u16* Cb, int Nout, int K, int relu) {
    dim3 g(Nout / 256, MPAD / 128);
    k_gemm<<<g, 256, 0, stream>>>(A, W, bias, Cf, Cb, Nout, K, relu);
  };

  // segment bookkeeping
  k_zero_counts<<<1, NGRAPH, 0, stream>>>(counts);
  k_count<<<blks(NTOK, 256), 256, 0, stream>>>(batch, counts);
  k_scan<<<1, 1, 0, stream>>>(counts, offsets);
  k_pos<<<blks(NTOK, 256), 256, 0, stream>>>(batch, offsets, pos);

  // scatter into padded f32 + bf16 mirror
  const size_t HN = (size_t)MPAD * DMODEL;
  k_zero_f32<<<blks(HN, 256), 256, 0, stream>>>(h, HN);
  k_scatter<<<blks((size_t)NTOK * DMODEL, 256), 256, 0, stream>>>(x, batch, pos, h);
  k_cvt_bf16<<<blks(HN, 256), 256, 0, stream>>>(h, hb, HN);

  // weights -> bf16
  k_cvt_bf16<<<blks((size_t)NLAYER*3*DMODEL*DMODEL,256),256,0,stream>>>(Wqkv, wq, (size_t)NLAYER*3*DMODEL*DMODEL);
  k_cvt_bf16<<<blks((size_t)NLAYER*DMODEL*DMODEL,256),256,0,stream>>>(Wo, wo, (size_t)NLAYER*DMODEL*DMODEL);
  k_cvt_bf16<<<blks((size_t)NLAYER*FFDIM*DMODEL,256),256,0,stream>>>(W1, w1, (size_t)NLAYER*FFDIM*DMODEL);
  k_cvt_bf16<<<blks((size_t)NLAYER*DMODEL*FFDIM,256),256,0,stream>>>(W2, w2, (size_t)NLAYER*DMODEL*FFDIM);
  k_cvt_bf16<<<blks((size_t)FFDIM*DMODEL,256),256,0,stream>>>(Wm1, wm1, (size_t)FFDIM*DMODEL);
  k_cvt_bf16<<<blks((size_t)DMODEL*FFDIM,256),256,0,stream>>>(Wm2, wm2, (size_t)DMODEL*FFDIM);

  const int VT_BLOCKS = NGRAPH * NHEAD * (DHEAD / 32) * (LMAX / 32);
  for (int l = 0; l < NLAYER; ++l) {
    gemm(hb, wq + (size_t)l*3*DMODEL*DMODEL, bqkv + l*3*DMODEL, nullptr, qkv, 3*DMODEL, DMODEL, 0);
    k_vtrans<<<VT_BLOCKS, 256, 0, stream>>>(qkv, vt);
    k_attn<<<NGRAPH * NHEAD * (LMAX / 16), 32, 0, stream>>>(qkv, vt, counts, hb);
    gemm(hb, wo + (size_t)l*DMODEL*DMODEL, bo + l*DMODEL, y, nullptr, DMODEL, DMODEL, 0);
    k_ln<<<MPAD, 256, 0, stream>>>(h, y, ln1w + l*DMODEL, ln1b + l*DMODEL, hb);
    gemm(hb, w1 + (size_t)l*FFDIM*DMODEL, b1 + l*FFDIM, nullptr, ffb, FFDIM, DMODEL, 1);
    gemm(ffb, w2 + (size_t)l*DMODEL*FFDIM, b2 + l*DMODEL, y, nullptr, DMODEL, FFDIM, 0);
    k_ln<<<MPAD, 256, 0, stream>>>(h, y, ln2w + l*DMODEL, ln2b + l*DMODEL, hb);
  }
  // final MLP + gather
  gemm(hb, wm1, bm1, nullptr, ffb, FFDIM, DMODEL, 1);
  gemm(ffb, wm2, bm2, y, nullptr, DMODEL, FFDIM, 0);
  k_gather<<<blks((size_t)NTOK * DMODEL, 256), 256, 0, stream>>>(y, batch, pos, outp);
}
